// OmDetTurboMultiscaleDeformableAttention_35656818491756
// MI455X (gfx1250) — matrix operand order repaired and checked by reading, not compiled, not run on hardware
//
#include <hip/hip_runtime.h>
#include <hip/hip_bf16.h>

typedef __attribute__((ext_vector_type(16))) _Float16 v16h;
typedef __attribute__((ext_vector_type(8)))  float    v8f;

#define B_DIM   8
#define Q_DIM   1000
#define D_MODEL 256
#define H_HEADS 8
#define L_LEV   4
#define P_PTS   4
#define S_TOTAL 13294   // 100*100 + 50*50 + 25*25 + 13*13

__device__ __constant__ int c_lvl_h[4]     = {100, 50, 25, 13};
__device__ __constant__ int c_lvl_w[4]     = {100, 50, 25, 13};
__device__ __constant__ int c_lvl_start[4] = {0, 10000, 12500, 13125};

// ---------------------------------------------------------------------------
// Pack a row-major f32 weight matrix B[256, N] into fragment-ordered f16:
// for n-tile t, k-block kb, lane l: 16 contiguous halfs (32 B) where element
// e maps to K = kb*32 + (l>>4)*16 + e and column n = t*16 + (l&15)
// (contiguous k order, same as the A fragment). One thread per (t,kb,lane).
// ---------------------------------------------------------------------------
__global__ void pack_b_f16_kernel(const float* __restrict__ Bm,
                                  _Float16* __restrict__ Bp, int N) {
  const int idx = blockIdx.x * blockDim.x + threadIdx.x;  // ((t*8+kb)*32)+lane
  const int total = (N / 16) * 8 * 32;
  if (idx >= total) return;
  const int lane = idx & 31;
  const int kb   = (idx >> 5) & 7;
  const int t    = idx >> 8;
  const int n     = t * 16 + (lane & 15);
  const int kbase = kb * 32 + (lane >> 4) * 16;
  _Float16* dst = Bp + (size_t)idx * 16;
#pragma unroll
  for (int e = 0; e < 16; ++e)
    dst[e] = (_Float16)Bm[(size_t)(kbase + e) * N + n];
}

// ---------------------------------------------------------------------------
// GEMM: C[M,N] = A[M,256] @ B[256,N] + bias, f32 A converted in-register to
// f16, B pre-packed f16 fragments, f32 WMMA accumulate.
// Block = 4 waves; wave w owns n-tiles [w*NT, w*NT+NT); 4*NT*16 == N.
// Per K-step: one A fragment (4x b128 + cvt), NT B fragments (each one
// aligned v16h = 2x b128, contiguous across the wave), NT v_wmma.
// ---------------------------------------------------------------------------
template <int NT>
__global__ void wmma_gemm_k256(const float* __restrict__ A,
                               const _Float16* __restrict__ Bp,
                               const float* __restrict__ bias,
                               float* __restrict__ C, int N) {
  const int lane   = threadIdx.x & 31;
  const int wave   = threadIdx.x >> 5;
  const int m_tile = blockIdx.x;
  const int nt0    = wave * NT;

  const int col = lane & 15;   // A-frag row / B,C-frag col
  const int hi  = lane >> 4;   // half-wave selector

  const float* __restrict__ Arow = A + (size_t)(m_tile * 16 + col) * 256;

  v8f acc[NT];
#pragma unroll
  for (int nt = 0; nt < NT; ++nt) acc[nt] = (v8f){};

#pragma unroll
  for (int kb = 0; kb < 8; ++kb) {
    v16h a;
#pragma unroll
    for (int e = 0; e < 16; ++e) {
      const int ka = kb * 32 + hi * 16 + e;
      a[e] = (_Float16)Arow[ka];
    }
#pragma unroll
    for (int nt = 0; nt < NT; ++nt) {
      const v16h b = *(const v16h*)(Bp +
          ((size_t)(((nt0 + nt) * 8 + kb) * 32) + lane) * 16);
      acc[nt] = __builtin_amdgcn_wmma_f32_16x16x32_f16(
          false, a, false, b, (short)0, acc[nt], false, false);
    }
  }

#pragma unroll
  for (int nt = 0; nt < NT; ++nt) {
    const float bv = bias[(nt0 + nt) * 16 + col];
#pragma unroll
    for (int r = 0; r < 8; ++r) {
      C[(size_t)(m_tile * 16 + hi * 8 + r) * N + (nt0 + nt) * 16 + col] =
          acc[nt][r] + bv;
    }
  }
}

// ---------------------------------------------------------------------------
// In-place softmax over rows of 16 (the L*P axis). One thread per (b,q,h) row.
// ---------------------------------------------------------------------------
__global__ void softmax16_kernel(float* __restrict__ aw, int rows) {
  const int r = blockIdx.x * blockDim.x + threadIdx.x;
  if (r >= rows) return;
  float* p = aw + (size_t)r * 16;
  float v[16];
  float mx = -3.402823e38f;
#pragma unroll
  for (int i = 0; i < 16; ++i) { v[i] = p[i]; mx = fmaxf(mx, v[i]); }
  float s = 0.f;
#pragma unroll
  for (int i = 0; i < 16; ++i) { v[i] = expf(v[i] - mx); s += v[i]; }
  const float inv = 1.f / s;
#pragma unroll
  for (int i = 0; i < 16; ++i) p[i] = v[i] * inv;
}

// ---------------------------------------------------------------------------
// Deformable bilinear sampling + attention-weighted sum.
// One wave per (b,q,h); lane = channel d (D=32 == wave32): each corner gather
// of value[b, s+idx, h, 0:32] is one coalesced 128B line. Bilinear math is
// wave-uniform.
// ---------------------------------------------------------------------------
__global__ void deform_sample_kernel(const float* __restrict__ value, // [B*S, 256]
                                     const float* __restrict__ ref,   // [B,Q,L,2]
                                     const float* __restrict__ off,   // [B*Q, 256] = (h,l,p,2)
                                     const float* __restrict__ aw,    // [B*Q, 128] = (h,l,p)
                                     float* __restrict__ mid) {       // [B*Q, 256]
  const int lane = threadIdx.x & 31;                       // channel d
  const int wave = threadIdx.x >> 5;
  const int g = blockIdx.x * (blockDim.x >> 5) + wave;     // (b,q,h) flat
  if (g >= B_DIM * Q_DIM * H_HEADS) return;

  const int h  = g % H_HEADS;
  const int bq = g / H_HEADS;
  const int b  = bq / Q_DIM;

  const float* __restrict__ offp = off + (size_t)bq * 256 + h * 32;
  const float* __restrict__ awp  = aw  + (size_t)bq * 128 + h * 16;
  const float* __restrict__ refp = ref + (size_t)bq * 8;

  float acc = 0.f;
#pragma unroll
  for (int l = 0; l < L_LEV; ++l) {
    const int hh = c_lvl_h[l];
    const int ww = c_lvl_w[l];
    const float* __restrict__ vbase =
        value + ((size_t)(b * S_TOTAL + c_lvl_start[l])) * D_MODEL + h * 32 + lane;
    const float rx = refp[l * 2 + 0];
    const float ry = refp[l * 2 + 1];
#pragma unroll
    for (int p = 0; p < P_PTS; ++p) {
      const float ox = offp[(l * P_PTS + p) * 2 + 0];
      const float oy = offp[(l * P_PTS + p) * 2 + 1];
      const float w8 = awp[l * P_PTS + p];
      const float x = (rx + ox / (float)ww) * (float)ww - 0.5f;
      const float y = (ry + oy / (float)hh) * (float)hh - 0.5f;
      const float x0f = floorf(x);
      const float y0f = floorf(y);
      const float fx = x - x0f;
      const float fy = y - y0f;
      const int x0 = (int)x0f;
      const int y0 = (int)y0f;
      float s = 0.f;
#pragma unroll
      for (int c = 0; c < 4; ++c) {
        const int dx = c & 1, dy = c >> 1;
        const int xi = x0 + dx, yi = y0 + dy;
        if (xi >= 0 && xi < ww && yi >= 0 && yi < hh) {   // wave-uniform branch
          const float wt = (dx ? fx : 1.f - fx) * (dy ? fy : 1.f - fy);
          s += wt * vbase[(size_t)(yi * ww + xi) * D_MODEL];
        }
      }
      acc += w8 * s;
    }
  }
  mid[(size_t)bq * D_MODEL + h * 32 + lane] = acc;
}

extern "C" void kernel_launch(void* const* d_in, const int* in_sizes, int n_in,
                              void* d_out, int out_size, void* d_ws, size_t ws_size,
                              hipStream_t stream) {
  const float* hidden  = (const float*)d_in[0];   // [B,Q,256]
  const float* encoder = (const float*)d_in[1];   // [B,S,256]
  const float* refpts  = (const float*)d_in[2];   // [B,Q,L,2]
  const float* W_value = (const float*)d_in[3];
  const float* b_value = (const float*)d_in[4];
  const float* W_off   = (const float*)d_in[5];
  const float* b_off   = (const float*)d_in[6];
  const float* W_attn  = (const float*)d_in[7];
  const float* b_attn  = (const float*)d_in[8];
  const float* W_out   = (const float*)d_in[9];
  const float* b_out   = (const float*)d_in[10];
  (void)in_sizes; (void)n_in; (void)out_size; (void)ws_size;

  float* out = (float*)d_out;                                     // [B*Q, 256]
  float* aw  = out + (size_t)B_DIM * Q_DIM * D_MODEL;             // [B*Q, 128]

  float* value = (float*)d_ws;                                    // [B*S, 256]
  float* off   = value + (size_t)B_DIM * S_TOTAL * D_MODEL;       // [B*Q, 256]
  float* mid   = off + (size_t)B_DIM * Q_DIM * D_MODEL;           // [B*Q, 256]
  _Float16* pk = (_Float16*)(mid + (size_t)B_DIM * Q_DIM * D_MODEL);
  _Float16* Bp_value = pk;                       // 256*256 halfs
  _Float16* Bp_off   = Bp_value + 256 * 256;     // 256*256
  _Float16* Bp_attn  = Bp_off   + 256 * 256;     // 256*128
  _Float16* Bp_out   = Bp_attn  + 256 * 128;     // 256*256

  const int M_val = B_DIM * S_TOTAL;   // 106352, multiple of 16
  const int M_hid = B_DIM * Q_DIM;     // 8000,   multiple of 16

  // 0) pack weight matrices into fragment-ordered f16 (tiny, L2-resident)
  pack_b_f16_kernel<<<(256 * 16 + 255) / 256, 256, 0, stream>>>(W_value, Bp_value, 256);
  pack_b_f16_kernel<<<(256 * 16 + 255) / 256, 256, 0, stream>>>(W_off,   Bp_off,   256);
  pack_b_f16_kernel<<<(128 * 16 + 255) / 256, 256, 0, stream>>>(W_attn,  Bp_attn,  128);
  pack_b_f16_kernel<<<(256 * 16 + 255) / 256, 256, 0, stream>>>(W_out,   Bp_out,   256);

  // 1) value = encoder @ W_value + b_value        [106352 x 256]
  wmma_gemm_k256<4><<<M_val / 16, 128, 0, stream>>>(encoder, Bp_value, b_value, value, 256);
  // 2) off = hidden @ W_off + b_off               [8000 x 256]
  wmma_gemm_k256<4><<<M_hid / 16, 128, 0, stream>>>(hidden, Bp_off, b_off, off, 256);
  // 3) attn logits = hidden @ W_attn + b_attn     [8000 x 128] -> aw region of d_out
  wmma_gemm_k256<2><<<M_hid / 16, 128, 0, stream>>>(hidden, Bp_attn, b_attn, aw, 128);
  // 4) softmax over L*P=16, in place              64000 rows
  softmax16_kernel<<<(B_DIM * Q_DIM * H_HEADS + 255) / 256, 256, 0, stream>>>(aw, B_DIM * Q_DIM * H_HEADS);
  // 5) bilinear sample + weighted sum -> mid      one wave per (b,q,h)
  deform_sample_kernel<<<(B_DIM * Q_DIM * H_HEADS) / 8, 256, 0, stream>>>(value, refpts, off, aw, mid);
  // 6) output = mid @ W_out + b_out               [8000 x 256] -> d_out
  wmma_gemm_k256<4><<<M_hid / 16, 128, 0, stream>>>(mid, Bp_out, b_out, out, 256);
}
